// PointNetPPXYZ_Schedmit_55456617726255
// MI455X (gfx1250) — compile-verified, matrix operand order checked by reading, not run on hardware
//
#include <hip/hip_runtime.h>
#include <math.h>

typedef _Float16 v16h __attribute__((ext_vector_type(16)));
typedef float    v8f  __attribute__((ext_vector_type(8)));

#define EPS_BN 1e-5f

// ---------------------------------------------------------------------------
// Weight packing: W (K x N, f32 row-major) -> per-lane WMMA B operand layout,
// f16, tiles of 32(K) x 16(N).  Packed index: ((kt*ntiles+nt)*32 + lane)*16 + p
// with true k = kt*32 + (lane>>4)*16 + p, n = nt*16 + (lane&15).  This matches
// the contiguous-K per-lane A load in the GEMM (permutations cancel).
// ---------------------------------------------------------------------------
__global__ void pack_w_kernel(const float* __restrict__ W, int K, int N,
                              int ktiles, int ntiles, _Float16* __restrict__ Bp)
{
    size_t id  = (size_t)blockIdx.x * 256 + threadIdx.x;
    size_t tot = (size_t)ktiles * ntiles * 512;
    if (id >= tot) return;
    int    p    = (int)(id & 15);
    int    lane = (int)((id >> 4) & 31);
    size_t tile = id >> 9;
    int    nt   = (int)(tile % ntiles);
    int    kt   = (int)(tile / ntiles);
    int    k    = kt * 32 + ((lane >> 4) << 4) + p;
    int    n    = nt * 16 + (lane & 15);
    float  val  = (k < K && n < N) ? W[(size_t)k * N + n] : 0.f;
    Bp[id] = (_Float16)val;
}

// ---------------------------------------------------------------------------
// Tiled WMMA GEMM + fused BN + ReLU epilogue.
//   C(M,N) = relu( (A(M,K)*W + b - m) * g * rsqrt(v+eps) + be )
// A: f16 row-major, lda = Kpad (multiple of 32).  Bp: packed weights.
// Block = 128 threads (4 waves).  Each wave computes a 16(M) x 64(N) strip:
// one A fragment load feeds 4 independent v_wmma per K-step (4x A reuse,
// hazard-free back-to-back wmma issue).  Grid = (ceil(mtiles/4), ntiles/4).
// ---------------------------------------------------------------------------
__global__ void gemm_bn_relu_kernel(const _Float16* __restrict__ A, int lda,
                                    const _Float16* __restrict__ Bp,
                                    int ktiles, int ntiles, int mtiles,
                                    const float* __restrict__ b,
                                    const float* __restrict__ g,
                                    const float* __restrict__ be,
                                    const float* __restrict__ m,
                                    const float* __restrict__ v,
                                    _Float16* __restrict__ O, int ldo)
{
    const int lane  = threadIdx.x & 31;
    const int wave  = threadIdx.x >> 5;
    const int mTile = blockIdx.x * 4 + wave;
    if (mTile >= mtiles) return;                 // uniform per wave: EXEC stays all-1s
    const int mBase = mTile * 16;
    const int nt0   = blockIdx.y * 4;

    const _Float16* aPtr = A + (size_t)(mBase + (lane & 15)) * lda + ((lane >> 4) << 4);
    const _Float16* bPtr = Bp + ((size_t)nt0 * 32 + lane) * 16;
    const size_t    bStride = (size_t)ntiles * 512;

    v8f acc0 = {}, acc1 = {}, acc2 = {}, acc3 = {};
    for (int kt = 0; kt < ktiles; ++kt) {
        v16h av = *(const v16h*)aPtr;
        v16h b0 = *(const v16h*)(bPtr);
        v16h b1 = *(const v16h*)(bPtr + 512);
        v16h b2 = *(const v16h*)(bPtr + 1024);
        v16h b3 = *(const v16h*)(bPtr + 1536);
        acc0 = __builtin_amdgcn_wmma_f32_16x16x32_f16(false, av, false, b0, (short)0, acc0, false, false);
        acc1 = __builtin_amdgcn_wmma_f32_16x16x32_f16(false, av, false, b1, (short)0, acc1, false, false);
        acc2 = __builtin_amdgcn_wmma_f32_16x16x32_f16(false, av, false, b2, (short)0, acc2, false, false);
        acc3 = __builtin_amdgcn_wmma_f32_16x16x32_f16(false, av, false, b3, (short)0, acc3, false, false);
        aPtr += 32;
        bPtr += bStride;
    }

    const int mOff = (lane >> 4) * 8;
    v8f accs[4] = {acc0, acc1, acc2, acc3};
#pragma unroll
    for (int j = 0; j < 4; ++j) {
        const int n = (nt0 + j) * 16 + (lane & 15);
        const float inv   = rsqrtf(v[n] + EPS_BN);
        const float scale = g[n] * inv;
        const float shift = (b[n] - m[n]) * scale + be[n];
#pragma unroll
        for (int i = 0; i < 8; ++i) {
            float val = accs[j][i] * scale + shift;
            val = fmaxf(val, 0.f);
            O[(size_t)(mBase + mOff + i) * ldo + n] = (_Float16)val;
        }
    }
}

// ---------------------------------------------------------------------------
// Top-32 nearest neighbours.  One block per (centroid); distances staged in
// LDS, then 32 rounds of tree arg-min.  grid = (M, B), block = 256.
// ---------------------------------------------------------------------------
#define TK_THREADS 256
__global__ void topk32_kernel(const float* __restrict__ pts, int N,
                              const float* __restrict__ q, int M,
                              int* __restrict__ outIdx)
{
    __shared__ float dist[8192];
    __shared__ float rv[TK_THREADS];
    __shared__ int   ri[TK_THREADS];

    const int b = blockIdx.y;
    const int c = blockIdx.x;
    const float qx = q[((size_t)b * M + c) * 3 + 0];
    const float qy = q[((size_t)b * M + c) * 3 + 1];
    const float qz = q[((size_t)b * M + c) * 3 + 2];
    const float* P = pts + (size_t)b * N * 3;

    for (int i = threadIdx.x; i < N; i += TK_THREADS) {
        float dx = P[(size_t)i * 3 + 0] - qx;
        float dy = P[(size_t)i * 3 + 1] - qy;
        float dz = P[(size_t)i * 3 + 2] - qz;
        dist[i] = dx * dx + dy * dy + dz * dz;
    }
    __syncthreads();

    for (int k = 0; k < 32; ++k) {
        float best = 3.4e38f; int bi = 0;
        for (int i = threadIdx.x; i < N; i += TK_THREADS) {
            float d = dist[i];
            if (d < best) { best = d; bi = i; }
        }
        rv[threadIdx.x] = best; ri[threadIdx.x] = bi;
        __syncthreads();
        for (int s = TK_THREADS / 2; s > 0; s >>= 1) {
            if (threadIdx.x < s && rv[threadIdx.x + s] < rv[threadIdx.x]) {
                rv[threadIdx.x] = rv[threadIdx.x + s];
                ri[threadIdx.x] = ri[threadIdx.x + s];
            }
            __syncthreads();
        }
        if (threadIdx.x == 0) {
            int w = ri[0];
            outIdx[((size_t)b * M + c) * 32 + k] = w;
            dist[w] = 3.4e38f;
        }
        __syncthreads();
    }
}

// centroid gather: out[b,c,:] = pts[b, idx[b,c], :]
__global__ void gather_cent_kernel(const float* __restrict__ pts, int N,
                                   const int* __restrict__ idx, int M, int B,
                                   float* __restrict__ out)
{
    int t = blockIdx.x * 256 + threadIdx.x;
    if (t >= B * M) return;
    int b = t / M;
    int p = idx[t];
    const float* s = pts + ((size_t)b * N + p) * 3;
    float* o = out + (size_t)t * 3;
    o[0] = s[0]; o[1] = s[1]; o[2] = s[2];
}

// SA1 feature build: rows = B*M*32, lda = 32; cols 0..2 rel-xyz, rest zero.
__global__ void build_feat1_kernel(const float* __restrict__ x, int N,
                                   const int* __restrict__ knn,
                                   const float* __restrict__ cent,
                                   int M, int B, _Float16* __restrict__ A0)
{
    int r = blockIdx.x * 256 + threadIdx.x;
    int total = B * M * 32;
    if (r >= total) return;
    int bc = r >> 5;
    int b  = bc / M;
    int p  = knn[r];
    const float* P = x + ((size_t)b * N + p) * 3;
    const float* C = cent + (size_t)bc * 3;
    _Float16* o = A0 + (size_t)r * 32;
    o[0] = (_Float16)(P[0] - C[0]);
    o[1] = (_Float16)(P[1] - C[1]);
    o[2] = (_Float16)(P[2] - C[2]);
#pragma unroll
    for (int i = 3; i < 32; ++i) o[i] = (_Float16)0.f;
}

// SA2 feature build: rows = B*32*32, lda = 160; cols 0..2 rel-xyz,
// 3..130 = l1 features of neighbour, rest zero.
__global__ void build_feat2_kernel(const float* __restrict__ cent1,
                                   const float* __restrict__ cent2,
                                   const int* __restrict__ knn2,
                                   const _Float16* __restrict__ l1,
                                   int B, _Float16* __restrict__ B0)
{
    int r = blockIdx.x * 256 + threadIdx.x;
    int total = B * 32 * 32;
    if (r >= total) return;
    int bc = r >> 5;          // b*32 + c2
    int b  = bc / 32;
    int p  = knn2[r];         // in [0,128)
    const float* P = cent1 + ((size_t)b * 128 + p) * 3;
    const float* C = cent2 + (size_t)bc * 3;
    _Float16* o = B0 + (size_t)r * 160;
    o[0] = (_Float16)(P[0] - C[0]);
    o[1] = (_Float16)(P[1] - C[1]);
    o[2] = (_Float16)(P[2] - C[2]);
    const _Float16* f = l1 + ((size_t)b * 128 + p) * 128;
    for (int i = 0; i < 128; ++i) o[3 + i] = f[i];
    for (int i = 131; i < 160; ++i) o[i] = (_Float16)0.f;
}

// SA3 concat: rows = B*32, lda = 288; cols 0..2 = l2_xyz, 3..258 = l2 feats.
__global__ void concat3_kernel(const float* __restrict__ cent2,
                               const _Float16* __restrict__ l2,
                               int B, _Float16* __restrict__ C0)
{
    int r = blockIdx.x * 256 + threadIdx.x;
    if (r >= B * 32) return;
    const float* C = cent2 + (size_t)r * 3;
    _Float16* o = C0 + (size_t)r * 288;
    o[0] = (_Float16)C[0]; o[1] = (_Float16)C[1]; o[2] = (_Float16)C[2];
    const _Float16* f = l2 + (size_t)r * 256;
    for (int i = 0; i < 256; ++i) o[3 + i] = f[i];
    for (int i = 259; i < 288; ++i) o[i] = (_Float16)0.f;
}

// max-pool over 32 consecutive rows per group
__global__ void pool32_kernel(const _Float16* __restrict__ in, int C,
                              _Float16* __restrict__ out, int G)
{
    int t = blockIdx.x * 256 + threadIdx.x;
    if (t >= G * C) return;
    int g = t / C, c = t % C;
    const _Float16* p = in + ((size_t)g * 32) * C + c;
    float mx = -3.4e38f;
    for (int j = 0; j < 32; ++j) {
        float v = (float)p[(size_t)j * C];
        mx = fmaxf(mx, v);
    }
    out[t] = (_Float16)mx;
}

// heads: v2 = l2norm(F2 @ Wy + by), v3 = l2norm(F2 @ Wz + bz)
__global__ void heads_kernel(const _Float16* __restrict__ F2,
                             const float* __restrict__ Wy, const float* __restrict__ by,
                             const float* __restrict__ Wz, const float* __restrict__ bz,
                             float* __restrict__ out)
{
    int b = threadIdx.x;
    if (b >= 32) return;
    float y0 = by[0], y1 = by[1], y2 = by[2];
    float z0 = bz[0], z1 = bz[1], z2 = bz[2];
    const _Float16* f = F2 + (size_t)b * 256;
    for (int k = 0; k < 256; ++k) {
        float x = (float)f[k];
        y0 += x * Wy[k * 3 + 0]; y1 += x * Wy[k * 3 + 1]; y2 += x * Wy[k * 3 + 2];
        z0 += x * Wz[k * 3 + 0]; z1 += x * Wz[k * 3 + 1]; z2 += x * Wz[k * 3 + 2];
    }
    float ny = fmaxf(sqrtf(y0 * y0 + y1 * y1 + y2 * y2), 1e-12f);
    float nz = fmaxf(sqrtf(z0 * z0 + z1 * z1 + z2 * z2), 1e-12f);
    out[b * 3 + 0] = y0 / ny; out[b * 3 + 1] = y1 / ny; out[b * 3 + 2] = y2 / ny;
    out[96 + b * 3 + 0] = z0 / nz; out[96 + b * 3 + 1] = z1 / nz; out[96 + b * 3 + 2] = z2 / nz;
}

// ---------------------------------------------------------------------------
// Host orchestration
// ---------------------------------------------------------------------------
static void mlp_layer(const _Float16* A, int M, int K, int Kpad, int N,
                      const float* W, const float* b, const float* g,
                      const float* be, const float* m, const float* v,
                      _Float16* O, _Float16* wpack, hipStream_t s)
{
    int ktiles = Kpad / 32;
    int ntiles = N / 16;
    int mtiles = M / 16;
    size_t tot = (size_t)ktiles * ntiles * 512;
    pack_w_kernel<<<(unsigned)((tot + 255) / 256), 256, 0, s>>>(W, K, N, ktiles, ntiles, wpack);
    dim3 grid((mtiles + 3) / 4, ntiles / 4);
    gemm_bn_relu_kernel<<<grid, 128, 0, s>>>(A, Kpad, wpack, ktiles, ntiles, mtiles,
                                             b, g, be, m, v, O, N);
}

extern "C" void kernel_launch(void* const* d_in, const int* in_sizes, int n_in,
                              void* d_out, int out_size, void* d_ws, size_t ws_size,
                              hipStream_t stream)
{
    (void)in_sizes; (void)n_in; (void)out_size; (void)ws_size;
    const int B = 32, N = 8192;

    // ---- input mapping (setup_inputs dict order, nested dict insertion order)
    const float* x = (const float*)d_in[0];
    // sa1 layers: [1..6], [7..12], [13..18]   (W,b,g,be,m,v)
    // sa2 layers: [19..24], [25..30], [31..36]
    // sa3 layers: [37..42], [43..48], [49..54]
    // fc1: 55 W, 56 b ; bn1: 57 g, 58 be, 59 m, 60 v
    // fc2: 61 W, 62 b ; bn2: 63 g, 64 be, 65 m, 66 v
    // head_y: 67 W, 68 b ; head_z: 69 W, 70 b
    const int* fps1 = (const int*)d_in[71];   // (32,128)
    const int* fps2 = (const int*)d_in[72];   // (32,32)
    float* out = (float*)d_out;

    // ---- workspace carve-up (256-byte aligned)
    char* ws = (char*)d_ws;
    size_t off = 0;
    auto carve = [&](size_t bytes) -> char* {
        char* p = ws + off;
        off += (bytes + 255) & ~(size_t)255;
        return p;
    };
    _Float16* P0    = (_Float16*)carve(16777216);            // ping  (max 131072*64*2)
    _Float16* P1    = (_Float16*)carve(33554432);            // pong  (max 131072*128*2)
    float*    cent1 = (float*)   carve(4096u * 3 * 4);       // (B,128,3)
    float*    cent2 = (float*)   carve(1024u * 3 * 4);       // (B,32,3)
    int*      idx1  = (int*)     carve(131072u * 4);         // (B,128,32)
    int*      idx2  = (int*)     carve(32768u * 4);          // (B,32,32)
    _Float16* l1    = (_Float16*)carve(4096u * 128 * 2);     // pooled SA1
    _Float16* l2    = (_Float16*)carve(1024u * 256 * 2);     // pooled SA2
    _Float16* C0    = (_Float16*)carve(1024u * 288 * 2);
    _Float16* C1    = (_Float16*)carve(1024u * 256 * 2);
    _Float16* C2    = (_Float16*)carve(1024u * 512 * 2);
    _Float16* C3    = (_Float16*)carve(1024u * 1024 * 2);
    _Float16* Gb    = (_Float16*)carve(32u * 1024 * 2);      // global feature
    _Float16* F1    = (_Float16*)carve(32u * 512 * 2);
    _Float16* F2    = (_Float16*)carve(32u * 256 * 2);
    _Float16* wpack = (_Float16*)carve(1024u * 512 * 2);     // reused per layer

    auto fw = [&](int i) { return (const float*)d_in[i]; };

    // ===== SA1 =====
    gather_cent_kernel<<<(B * 128 + 255) / 256, 256, 0, stream>>>(x, N, fps1, 128, B, cent1);
    topk32_kernel<<<dim3(128, B), TK_THREADS, 0, stream>>>(x, N, cent1, 128, idx1);
    build_feat1_kernel<<<(B * 128 * 32 + 255) / 256, 256, 0, stream>>>(x, N, idx1, cent1, 128, B, P0);
    mlp_layer(P0, 131072,   3,  32,  64, fw(1),  fw(2),  fw(3),  fw(4),  fw(5),  fw(6),  P1, wpack, stream);
    mlp_layer(P1, 131072,  64,  64,  64, fw(7),  fw(8),  fw(9),  fw(10), fw(11), fw(12), P0, wpack, stream);
    mlp_layer(P0, 131072,  64,  64, 128, fw(13), fw(14), fw(15), fw(16), fw(17), fw(18), P1, wpack, stream);
    pool32_kernel<<<(4096 * 128 + 255) / 256, 256, 0, stream>>>(P1, 128, l1, 4096);

    // ===== SA2 =====
    gather_cent_kernel<<<(B * 32 + 255) / 256, 256, 0, stream>>>(cent1, 128, fps2, 32, B, cent2);
    topk32_kernel<<<dim3(32, B), TK_THREADS, 0, stream>>>(cent1, 128, cent2, 32, idx2);
    build_feat2_kernel<<<(B * 32 * 32 + 255) / 256, 256, 0, stream>>>(cent1, cent2, idx2, l1, B, P0);
    mlp_layer(P0, 32768, 131, 160, 128, fw(19), fw(20), fw(21), fw(22), fw(23), fw(24), P1, wpack, stream);
    mlp_layer(P1, 32768, 128, 128, 128, fw(25), fw(26), fw(27), fw(28), fw(29), fw(30), P0, wpack, stream);
    mlp_layer(P0, 32768, 128, 128, 256, fw(31), fw(32), fw(33), fw(34), fw(35), fw(36), P1, wpack, stream);
    pool32_kernel<<<(1024 * 256 + 255) / 256, 256, 0, stream>>>(P1, 256, l2, 1024);

    // ===== SA3 =====
    concat3_kernel<<<(B * 32 + 255) / 256, 256, 0, stream>>>(cent2, l2, B, C0);
    mlp_layer(C0, 1024, 259, 288,  256, fw(37), fw(38), fw(39), fw(40), fw(41), fw(42), C1, wpack, stream);
    mlp_layer(C1, 1024, 256, 256,  512, fw(43), fw(44), fw(45), fw(46), fw(47), fw(48), C2, wpack, stream);
    mlp_layer(C2, 1024, 512, 512, 1024, fw(49), fw(50), fw(51), fw(52), fw(53), fw(54), C3, wpack, stream);
    pool32_kernel<<<(32 * 1024 + 255) / 256, 256, 0, stream>>>(C3, 1024, Gb, 32);

    // ===== FC head =====
    mlp_layer(Gb, 32, 1024, 1024, 512, fw(55), fw(56), fw(57), fw(58), fw(59), fw(60), F1, wpack, stream);
    mlp_layer(F1, 32,  512,  512, 256, fw(61), fw(62), fw(63), fw(64), fw(65), fw(66), F2, wpack, stream);
    heads_kernel<<<1, 32, 0, stream>>>(F2, fw(67), fw(68), fw(69), fw(70), out);
}
